// GPSA_80161269613480
// MI455X (gfx1250) — matrix operand order repaired and checked by reading, not compiled, not run on hardware
//
#include <hip/hip_runtime.h>
#include <hip/hip_bf16.h>
#include <stdint.h>

// ---- CDNA5 WMMA types -------------------------------------------------------
typedef __bf16 bf16;
typedef bf16     v16bf __attribute__((ext_vector_type(16)));
typedef bf16     v2bf  __attribute__((ext_vector_type(2)));
typedef float    v8f   __attribute__((ext_vector_type(8)));
typedef unsigned u32x4 __attribute__((ext_vector_type(4)));

#define Bb   64
#define Nn   300
#define Cc   256
#define Hh   8
#define HD   32
#define NP   320            // K-dim of attn*V padded to multiple of 32
#define Mtot (Bb * Nn)      // 19200

union Frag {
  v16bf v;
  unsigned short s[16];
  u32x4 u[2];
};

// scalar f32 -> bf16 (RNE)
__device__ __forceinline__ unsigned short f2bf(float f) {
  union { float f; unsigned u; } x; x.f = f;
  unsigned u = x.u;
  u += 0x7FFFu + ((u >> 16) & 1u);
  return (unsigned short)(u >> 16);
}

// packed pair f32x2 -> bf16x2 (v_cvt_pk_bf16_f32 on gfx1250 when available)
__device__ __forceinline__ unsigned pk2bf(float lo, float hi) {
#if __has_builtin(__builtin_amdgcn_cvt_pk_bf16_f32)
  union { v2bf v; unsigned u; } c;
  c.v = __builtin_amdgcn_cvt_pk_bf16_f32(lo, hi);
  return c.u;
#else
  return (unsigned)f2bf(lo) | ((unsigned)f2bf(hi) << 16);
#endif
}

// ---------------------------------------------------------------------------
// Kernel 0: one-time f32 -> bf16 conversion (x, Wqk||Wv, Wproj).
// 8 elements / thread, packed converts, 16B stores.
// ---------------------------------------------------------------------------
__global__ __launch_bounds__(256)
void cvt_kernel(const float* __restrict__ in, unsigned short* __restrict__ out,
                int n) {
  const int i = (blockIdx.x * 256 + threadIdx.x) * 8;
  if (i >= n) return;
  const float4 f0 = *(const float4*)(in + i);
  const float4 f1 = *(const float4*)(in + i + 4);
  union { u32x4 v; unsigned u[4]; } o;
  o.u[0] = pk2bf(f0.x, f0.y);
  o.u[1] = pk2bf(f0.z, f0.w);
  o.u[2] = pk2bf(f1.x, f1.y);
  o.u[3] = pk2bf(f1.z, f1.w);
  *(u32x4*)(out + i) = o.v;
}

// ---------------------------------------------------------------------------
// Kernel 1: fused QKV projection.  out = x @ [Wqk;Wv]^T,  M=19200 N=768 K=256.
// Each wave: 32x32 block (2x2 WMMA tiles, A/B fragments reused 2x).
// Stores q,k as bf16 [B,H,N,32]; v transposed bf16 [B,H,32,NP].
// ---------------------------------------------------------------------------
__device__ __forceinline__ void store_qkv(unsigned short* __restrict__ qb,
                                          unsigned short* __restrict__ kb,
                                          unsigned short* __restrict__ vT,
                                          int m, int n, float val) {
  const int bidx = m / Nn, j = m - bidx * Nn;
  const unsigned short bv = f2bf(val);
  if (n < Cc) {                               // q
    const int h = n >> 5, d = n & 31;
    qb[(((size_t)(bidx * Hh + h)) * Nn + j) * HD + d] = bv;
  } else if (n < 2 * Cc) {                    // k
    const int n2 = n - Cc, h = n2 >> 5, d = n2 & 31;
    kb[(((size_t)(bidx * Hh + h)) * Nn + j) * HD + d] = bv;
  } else {                                    // v, stored transposed [B,H,32,NP]
    const int n2 = n - 2 * Cc, h = n2 >> 5, d = n2 & 31;
    vT[(((size_t)(bidx * Hh + h)) * HD + d) * NP + j] = bv;
  }
}

__global__ __launch_bounds__(256)
void qkv_proj_kernel(const unsigned short* __restrict__ xb,   // [19200,256] bf16
                     const unsigned short* __restrict__ wb,   // [768,256]  bf16
                     unsigned short* __restrict__ qb,
                     unsigned short* __restrict__ kb,
                     unsigned short* __restrict__ vT) {
  const int wave = threadIdx.x >> 5;
  const int lane = threadIdx.x & 31;
  const int half = lane >> 4;
  const int r    = lane & 15;

  const int m0 = blockIdx.x * 32;
  const int n0 = (blockIdx.y * 8 + wave) * 32;     // 24 n-tiles of 32

  const unsigned short* a0p = xb + (size_t)(m0 + r)      * Cc;
  const unsigned short* a1p = xb + (size_t)(m0 + 16 + r) * Cc;
  const unsigned short* b0p = wb + (size_t)(n0 + r)      * Cc + 16 * half;
  const unsigned short* b1p = wb + (size_t)(n0 + 16 + r) * Cc + 16 * half;

  v8f cc[2][2] = {};
#pragma unroll
  for (int k0 = 0; k0 < Cc; k0 += 32) {
    Frag a0, a1, b0, b1;
    a0.u[0] = *(const u32x4*)(a0p + k0 + 16 * half);
    a0.u[1] = *(const u32x4*)(a0p + k0 + 8 + 16 * half);
    a1.u[0] = *(const u32x4*)(a1p + k0 + 16 * half);
    a1.u[1] = *(const u32x4*)(a1p + k0 + 8 + 16 * half);
    b0.v = *(const v16bf*)(b0p + k0);
    b1.v = *(const v16bf*)(b1p + k0);
    cc[0][0] = __builtin_amdgcn_wmma_f32_16x16x32_bf16(false, a0.v, false, b0.v, (short)0, cc[0][0], false, false);
    cc[0][1] = __builtin_amdgcn_wmma_f32_16x16x32_bf16(false, a0.v, false, b1.v, (short)0, cc[0][1], false, false);
    cc[1][0] = __builtin_amdgcn_wmma_f32_16x16x32_bf16(false, a1.v, false, b0.v, (short)0, cc[1][0], false, false);
    cc[1][1] = __builtin_amdgcn_wmma_f32_16x16x32_bf16(false, a1.v, false, b1.v, (short)0, cc[1][1], false, false);
  }
#pragma unroll
  for (int ti = 0; ti < 2; ++ti)
#pragma unroll
    for (int tj = 0; tj < 2; ++tj)
#pragma unroll
      for (int vv = 0; vv < 8; ++vv)
        store_qkv(qb, kb, vT,
                  m0 + ti * 16 + vv + 8 * half,
                  n0 + tj * 16 + r, cc[ti][tj][vv]);
}

// ---------------------------------------------------------------------------
// Kernel 2: positional softmax, batch-independent: pos[h,i,j],  [H,300,300] f32
// ---------------------------------------------------------------------------
__global__ __launch_bounds__(128)
void pos_kernel(const float* __restrict__ Wpos,
                const float* __restrict__ bpos,
                float* __restrict__ pos) {
  const int row = blockIdx.x;          // h*300 + i
  const int h = row / Nn, i = row % Nn;
  const float w0 = Wpos[h * 3 + 0], w2 = Wpos[h * 3 + 2], bb = bpos[h];
  __shared__ float red[128];
  const int t = threadIdx.x;
  float* prow = pos + (size_t)row * Nn;

  float mx = -3.0e38f;
  for (int j = t; j < Nn; j += 128) {
    const float d = (float)(j - i);
    const float val = w0 * d + w2 * d * d + bb;
    prow[j] = val;
    mx = fmaxf(mx, val);
  }
  red[t] = mx; __syncthreads();
  for (int s = 64; s > 0; s >>= 1) { if (t < s) red[t] = fmaxf(red[t], red[t + s]); __syncthreads(); }
  mx = red[0]; __syncthreads();

  float sum = 0.f;
  for (int j = t; j < Nn; j += 128) { const float e = __expf(prow[j] - mx); prow[j] = e; sum += e; }
  red[t] = sum; __syncthreads();
  for (int s = 64; s > 0; s >>= 1) { if (t < s) red[t] += red[t + s]; __syncthreads(); }
  const float inv = 1.0f / red[0];
  for (int j = t; j < Nn; j += 128) prow[j] *= inv;
}

// ---------------------------------------------------------------------------
// Kernel 3: fused attention for one (b,h) and a 16-row tile.
//   S = scale * q k^T (WMMA) -> LDS; softmax + gate-blend with pos + renorm;
//   attn (bf16, zero-padded to K=320) @ v  (WMMA on vT) -> Oh [B,N,C] bf16.
// ---------------------------------------------------------------------------
__global__ __launch_bounds__(128)
void attn_kernel(const unsigned short* __restrict__ qb,
                 const unsigned short* __restrict__ kb,
                 const unsigned short* __restrict__ vT,
                 const float* __restrict__ pos,
                 const float* __restrict__ gating,
                 unsigned short* __restrict__ Oh) {
  __shared__ float          S[16][NP];    // 20.0 KB
  __shared__ unsigned short At[16][NP];   // 10.0 KB
  __shared__ float          red[16][8];

  const int wave = threadIdx.x >> 5;
  const int lane = threadIdx.x & 31;
  const int half = lane >> 4;
  const int r    = lane & 15;

  const int m0 = blockIdx.x * 16;        // row tile (19 tiles cover 300)
  const int bh = blockIdx.y;             // b*H + h
  const int h  = bh & 7;
  const int b  = bh >> 3;

  // ---- phase 1: S = scale * q k^T ----
  const int qrow = (m0 + r < Nn) ? (m0 + r) : (Nn - 1);
  const unsigned short* qp = qb + (((size_t)bh) * Nn + qrow) * HD;
  Frag a;
  a.u[0] = *(const u32x4*)(qp + 16 * half);
  a.u[1] = *(const u32x4*)(qp + 8 + 16 * half);

  const float scale = 0.17677669529663687f;    // 1/sqrt(32)
  for (int ct = wave; ct < 19; ct += 4) {      // wave-uniform loop: EXEC all-1s
    const int jn0  = ct * 16;
    const int krow = (jn0 + r < Nn) ? (jn0 + r) : (Nn - 1);
    Frag bf;
    bf.v = *(const v16bf*)(kb + (((size_t)bh) * Nn + krow) * HD + 16 * half);
    v8f c = {};
    c = __builtin_amdgcn_wmma_f32_16x16x32_bf16(false, a.v, false, bf.v,
                                                (short)0, c, false, false);
#pragma unroll
    for (int vv = 0; vv < 8; ++vv)
      S[vv + 8 * half][jn0 + r] = c[vv] * scale;
  }
  // prefetch the v^T rows phase 3 will read, overlapping the softmax below
  if (wave < 2)
    __builtin_prefetch(vT + (((size_t)bh) * HD + wave * 16 + r) * NP, 0, 1);
  __syncthreads();

  // ---- phase 2: softmax + gated blend with positional scores ----
  const int row = threadIdx.x >> 3;   // 0..15
  const int sub = threadIdx.x & 7;
  const int ig  = m0 + row;
  const float g = 1.0f / (1.0f + __expf(-gating[h]));

  float mx = -3.0e38f;
  for (int j = sub; j < Nn; j += 8) mx = fmaxf(mx, S[row][j]);
  red[row][sub] = mx; __syncthreads();
  if (sub == 0) {
    float m2 = red[row][0];
#pragma unroll
    for (int t = 1; t < 8; ++t) m2 = fmaxf(m2, red[row][t]);
    red[row][0] = m2;
  }
  __syncthreads();
  mx = red[row][0]; __syncthreads();

  float sum = 0.f;
  for (int j = sub; j < Nn; j += 8) {
    const float e = __expf(S[row][j] - mx);
    S[row][j] = e; sum += e;
  }
  red[row][sub] = sum; __syncthreads();
  if (sub == 0) {
    float s2 = 0.f;
#pragma unroll
    for (int t = 0; t < 8; ++t) s2 += red[row][t];
    red[row][0] = s2;
  }
  __syncthreads();
  const float invp = 1.0f / red[row][0]; __syncthreads();

  const float* prow = pos + (((size_t)h) * Nn + (ig < Nn ? ig : Nn - 1)) * Nn;
  float tsum = 0.f;
  for (int j = sub; j < Nn; j += 8) {
    const float av = (1.0f - g) * S[row][j] * invp + g * prow[j];
    S[row][j] = av; tsum += av;
  }
  red[row][sub] = tsum; __syncthreads();
  if (sub == 0) {
    float s2 = 0.f;
#pragma unroll
    for (int t = 0; t < 8; ++t) s2 += red[row][t];
    red[row][0] = s2;
  }
  __syncthreads();
  const float invt = 1.0f / red[row][0];
  // pair-packed bf16 requantize of the attention row (zeros in the K-pad)
  for (int j0 = sub * 2; j0 < NP; j0 += 16) {
    const float a0 = (j0     < Nn) ? S[row][j0]     * invt : 0.0f;
    const float a1 = (j0 + 1 < Nn) ? S[row][j0 + 1] * invt : 0.0f;
    *(unsigned*)(&At[row][j0]) = pk2bf(a0, a1);
  }
  __syncthreads();

  // ---- phase 3: O = attn @ v  (K = 320, 10 WMMA steps; waves 0,1) ----
  if (wave < 2) {                          // wave-uniform branch
    const int n0 = wave * 16;              // head-dim tile (0 or 16)
    v8f c = {};
#pragma unroll
    for (int kk = 0; kk < NP; kk += 32) {
      Frag aa, bb;
      aa.u[0] = *(const u32x4*)(&At[r][kk + 16 * half]);
      aa.u[1] = *(const u32x4*)(&At[r][kk + 8 + 16 * half]);
      bb.v    = *(const v16bf*)(vT + (((size_t)bh) * HD + n0 + r) * NP
                                   + kk + 16 * half);
      c = __builtin_amdgcn_wmma_f32_16x16x32_bf16(false, aa.v, false, bb.v,
                                                  (short)0, c, false, false);
    }
#pragma unroll
    for (int vv = 0; vv < 8; ++vv) {
      const int j = m0 + vv + 8 * half;
      if (j < Nn)
        Oh[(((size_t)b) * Nn + j) * Cc + h * HD + n0 + r] = f2bf(c[vv]);
    }
  }
}

// ---------------------------------------------------------------------------
// Kernel 4: output projection  out = Oh @ Wproj^T + bproj   (M=19200 N=K=256)
// Each wave: 32x32 block (2x2 WMMA tiles); 8 waves cover all N=256.
// ---------------------------------------------------------------------------
__global__ __launch_bounds__(256)
void proj_kernel(const unsigned short* __restrict__ Oh,
                 const unsigned short* __restrict__ wpb,   // [256,256] bf16
                 const float* __restrict__ bproj,
                 float* __restrict__ out) {
  const int wave = threadIdx.x >> 5;
  const int lane = threadIdx.x & 31;
  const int half = lane >> 4;
  const int r    = lane & 15;

  const int m0 = blockIdx.x * 32;
  const int n0 = wave * 32;

  const unsigned short* a0p = Oh  + (size_t)(m0 + r)      * Cc;
  const unsigned short* a1p = Oh  + (size_t)(m0 + 16 + r) * Cc;
  const unsigned short* b0p = wpb + (size_t)(n0 + r)      * Cc + 16 * half;
  const unsigned short* b1p = wpb + (size_t)(n0 + 16 + r) * Cc + 16 * half;

  v8f cc[2][2] = {};
#pragma unroll
  for (int k0 = 0; k0 < Cc; k0 += 32) {
    Frag a0, a1, b0, b1;
    a0.u[0] = *(const u32x4*)(a0p + k0 + 16 * half);
    a0.u[1] = *(const u32x4*)(a0p + k0 + 8 + 16 * half);
    a1.u[0] = *(const u32x4*)(a1p + k0 + 16 * half);
    a1.u[1] = *(const u32x4*)(a1p + k0 + 8 + 16 * half);
    b0.v = *(const v16bf*)(b0p + k0);
    b1.v = *(const v16bf*)(b1p + k0);
    cc[0][0] = __builtin_amdgcn_wmma_f32_16x16x32_bf16(false, a0.v, false, b0.v, (short)0, cc[0][0], false, false);
    cc[0][1] = __builtin_amdgcn_wmma_f32_16x16x32_bf16(false, a0.v, false, b1.v, (short)0, cc[0][1], false, false);
    cc[1][0] = __builtin_amdgcn_wmma_f32_16x16x32_bf16(false, a1.v, false, b0.v, (short)0, cc[1][0], false, false);
    cc[1][1] = __builtin_amdgcn_wmma_f32_16x16x32_bf16(false, a1.v, false, b1.v, (short)0, cc[1][1], false, false);
  }
#pragma unroll
  for (int ti = 0; ti < 2; ++ti)
#pragma unroll
    for (int tj = 0; tj < 2; ++tj) {
      const int n = n0 + tj * 16 + r;
      const float bia = bproj[n];
#pragma unroll
      for (int vv = 0; vv < 8; ++vv) {
        const int m = m0 + ti * 16 + vv + 8 * half;
        out[(size_t)m * Cc + n] = cc[ti][tj][vv] + bia;
      }
    }
}

// ---------------------------------------------------------------------------
extern "C" void kernel_launch(void* const* d_in, const int* in_sizes, int n_in,
                              void* d_out, int out_size, void* d_ws, size_t ws_size,
                              hipStream_t stream) {
  (void)in_sizes; (void)n_in; (void)out_size; (void)ws_size;
  const float* x      = (const float*)d_in[0];
  const float* Wqk    = (const float*)d_in[1];
  const float* Wv     = (const float*)d_in[2];
  const float* Wpos   = (const float*)d_in[3];
  const float* bpos   = (const float*)d_in[4];
  const float* Wproj  = (const float*)d_in[5];
  const float* bproj  = (const float*)d_in[6];
  const float* gating = (const float*)d_in[7];
  float* out = (float*)d_out;

  // workspace layout (bytes)
  char* ws = (char*)d_ws;
  const size_t SZ_XB = (size_t)Mtot * Cc * 2;           //  9,830,400
  const size_t SZ_WB = (size_t)768 * Cc * 2;            //    393,216
  const size_t SZ_WP = (size_t)Cc * Cc * 2;             //    131,072
  const size_t SZ_QK = (size_t)Bb * Hh * Nn * HD * 2;   //  9,830,400 each
  const size_t SZ_VT = (size_t)Bb * Hh * HD * NP * 2;   // 10,485,760
  const size_t SZ_OH = (size_t)Mtot * Cc * 2;           //  9,830,400
  size_t off = 0;
  unsigned short* xb  = (unsigned short*)(ws + off); off += SZ_XB;
  unsigned short* wb  = (unsigned short*)(ws + off); off += SZ_WB;
  unsigned short* wpb = (unsigned short*)(ws + off); off += SZ_WP;
  unsigned short* qb  = (unsigned short*)(ws + off); off += SZ_QK;
  unsigned short* kb  = (unsigned short*)(ws + off); off += SZ_QK;
  unsigned short* vT  = (unsigned short*)(ws + off); off += SZ_VT;
  unsigned short* Oh  = (unsigned short*)(ws + off); off += SZ_OH;
  float*          pos = (float*)(ws + off);

  // one-time bf16 conversion (elementwise, packed cvt)
  cvt_kernel<<<dim3(Mtot * Cc / 2048), 256, 0, stream>>>(x, xb, Mtot * Cc);
  cvt_kernel<<<dim3(512 * Cc / 2048), 256, 0, stream>>>(Wqk, wb, 512 * Cc);
  cvt_kernel<<<dim3(256 * Cc / 2048), 256, 0, stream>>>(Wv, wb + 512 * Cc, 256 * Cc);
  cvt_kernel<<<dim3(256 * Cc / 2048), 256, 0, stream>>>(Wproj, wpb, 256 * Cc);

  qkv_proj_kernel<<<dim3(Mtot / 32, 3), 256, 0, stream>>>(xb, wb, qb, kb, vT);
  pos_kernel<<<dim3(Hh * Nn), 128, 0, stream>>>(Wpos, bpos, pos);
  attn_kernel<<<dim3(19, Bb * Hh), 128, 0, stream>>>(qb, kb, vT, pos, gating, Oh);
  proj_kernel<<<dim3(Mtot / 32), 256, 0, stream>>>(Oh, wpb, bproj, out);
}
